// SparseMoE_89635967468144
// MI455X (gfx1250) — matrix operand order repaired and checked by reading, not compiled, not run on hardware
//
#include <hip/hip_runtime.h>
#include <hip/hip_bf16.h>
#include <math.h>

// ---------------------------------------------------------------------------
// Sparse MoE for MI455X (gfx1250, wave32, WMMA 16x16x32 bf16, async-LDS copy)
// ---------------------------------------------------------------------------
#define N_TOK 4096
#define DIM   1024
#define HID   4096
#define NE    8

typedef unsigned short u16;
typedef unsigned int   u32;

typedef __attribute__((ext_vector_type(16))) __bf16 bf16x16;
typedef __attribute__((ext_vector_type(8)))  float  f32x8;

union FragBF { bf16x16 v; uint4 q[2]; };

__device__ __forceinline__ u16 f2bf(float f) {
    u32 u = __float_as_uint(f);
    u32 r = u + 0x7FFFu + ((u >> 16) & 1u);   // round-to-nearest-even
    return (u16)(r >> 16);
}

// pack two fp32 -> packed bf16 pair. The clang builtin is absent on this
// toolchain (verified via __has_builtin in a previous round), but the ISA has
// V_CVT_PK_BF16_F32 -- one VALU op replaces a ~12-op manual RNE sequence.
__device__ __forceinline__ u32 pack_bf16(float a, float b) {
    u32 d;
    asm("v_cvt_pk_bf16_f32 %0, %1, %2" : "=v"(d) : "v"(a), "v"(b));
    return d;
}

// 16-byte global -> LDS async copy (ASYNCcnt path). Inline asm: the clang
// builtin's pointer types are toolchain-specific; asm bypasses the frontend.
__device__ __forceinline__ void async_cp16(const void* g, void* l) {
    u32 laddr = (u32)(uintptr_t)l;   // low 32 bits of generic addr = LDS offset
    asm volatile("global_load_async_to_lds_b128 %0, %1, off"
                 :: "v"(laddr), "v"(g) : "memory");
}

#if __has_builtin(__builtin_amdgcn_s_wait_asynccnt)
#define ASYNC_WAIT0() __builtin_amdgcn_s_wait_asynccnt(0)
#else
#define ASYNC_WAIT0() asm volatile("s_wait_asynccnt 0x0" ::: "memory")
#endif

__device__ __forceinline__ float gelu_exact(float v) {
    return 0.5f * v * (1.0f + erff(v * 0.70710678118654752f));
}

// --------------------------- tiny utility kernels ---------------------------
__global__ void moe_zero_ctr(int* ctr) {           // cnt[8] | imp[8] | load[8]
    if (threadIdx.x < 24) ctr[threadIdx.x] = 0;
}

__global__ void moe_cvt_bf16(const float* __restrict__ x, u16* __restrict__ o) {
    int i = (blockIdx.x * blockDim.x + threadIdx.x) * 4;
    float4 v = *(const float4*)(x + i);
    uint2 p;
    p.x = pack_bf16(v.x, v.y);
    p.y = pack_bf16(v.z, v.w);
    *(uint2*)(o + i) = p;
}

// ------------------------------- gating -------------------------------------
__global__ __launch_bounds__(256) void moe_gating(
    const float* __restrict__ x, const float* __restrict__ gw,
    const float* __restrict__ gb,
    int* __restrict__ cnt, float* __restrict__ imp, float* __restrict__ loadf,
    int* __restrict__ list, float* __restrict__ cw)
{
    __shared__ float blkImp[NE];
    __shared__ float blkLoad[NE];
    int tid = threadIdx.x, lane = tid & 31, wave = tid >> 5;
    if (tid < NE) { blkImp[tid] = 0.f; blkLoad[tid] = 0.f; }
    __syncthreads();

    int token = blockIdx.x * 8 + wave;
    float s[NE];
#pragma unroll
    for (int i = 0; i < NE; i++) s[i] = 0.f;
    const float* xr = x + (size_t)token * DIM;
    for (int d = lane; d < DIM; d += 32) {
        float xv = xr[d];
        const float* g = gw + (size_t)d * NE;
        float4 g0 = *(const float4*)g;
        float4 g1 = *(const float4*)(g + 4);
        s[0] += xv * g0.x; s[1] += xv * g0.y; s[2] += xv * g0.z; s[3] += xv * g0.w;
        s[4] += xv * g1.x; s[5] += xv * g1.y; s[6] += xv * g1.z; s[7] += xv * g1.w;
    }
#pragma unroll
    for (int i = 0; i < NE; i++)
#pragma unroll
        for (int off = 16; off >= 1; off >>= 1)
            s[i] += __shfl_xor(s[i], off, 32);

    if (lane == 0) {
        float lg[NE];
#pragma unroll
        for (int i = 0; i < NE; i++) lg[i] = s[i] + gb[i];
        float mx = lg[0];
#pragma unroll
        for (int i = 1; i < NE; i++) mx = fmaxf(mx, lg[i]);
        float ex[NE], se = 0.f;
#pragma unroll
        for (int i = 0; i < NE; i++) { ex[i] = __expf(lg[i] - mx); se += ex[i]; }
        float inv = 1.f / se;
#pragma unroll
        for (int i = 0; i < NE; i++) atomicAdd(&blkImp[i], ex[i] * inv);
        int i0 = 0;
#pragma unroll
        for (int i = 1; i < NE; i++) if (lg[i] > lg[i0]) i0 = i;
        int i1 = (i0 == 0) ? 1 : 0;
#pragma unroll
        for (int i = 0; i < NE; i++) if (i != i0 && lg[i] > lg[i1]) i1 = i;
        float d1 = __expf(lg[i1] - lg[i0]);
        float w0 = 1.f / (1.f + d1);
        float w1 = 1.f - w0;
        int p0 = atomicAdd(&cnt[i0], 1);
        list[i0 * N_TOK + p0] = token; cw[i0 * N_TOK + p0] = w0;
        int p1 = atomicAdd(&cnt[i1], 1);
        list[i1 * N_TOK + p1] = token; cw[i1 * N_TOK + p1] = w1;
        atomicAdd(&blkLoad[i0], 1.f);
        atomicAdd(&blkLoad[i1], 1.f);
    }
    __syncthreads();
    if (tid < NE) {
        atomicAdd(&imp[tid],   blkImp[tid]);
        atomicAdd(&loadf[tid], blkLoad[tid]);
    }
}

__global__ void moe_finalize(const int* __restrict__ cnt,
                             const float* __restrict__ imp,
                             const float* __restrict__ loadf,
                             int* __restrict__ offs, float* __restrict__ out) {
    if (threadIdx.x == 0) {
        float si = 0.f, sl = 0.f;
        for (int e = 0; e < NE; e++) { si += imp[e]; sl += loadf[e]; }
        float aux = 0.f;
        for (int e = 0; e < NE; e++) {
            float a = imp[e]   / (si + 1e-8f);
            float b = loadf[e] / (sl + 1e-8f);
            aux += (a - b) * (a - b);
        }
        out[(size_t)N_TOK * DIM] = aux;
        int o = 0;
        for (int e = 0; e < NE; e++) { offs[e] = o; o += cnt[e]; }
    }
}

// ------------------------------ WMMA GEMM -----------------------------------
// Block tile C[128,128], K stepped by 32, double-buffered LDS.
// 8 waves as 4(M)x2(N); each wave computes 32x64 via 8 f32x8 accumulators.
// MODE 0: expert FFN1 : A=x_bf gathered,  out = gelu(.+b1) -> bf16 h slab
// MODE 1: shared FFN1 : A=x_bf,           out = gelu(.+b1) -> bf16 sh
// MODE 2: expert FFN2 : A=h_bf slab,      atomicAdd(out[token], cw*(.+b2))
// MODE 3: shared FFN2 : A=sh_bf,          out[row] = . + b2  (initializes d_out)
template<int MODE>
__global__ __launch_bounds__(256) void moe_gemm(
    const u16*  __restrict__ A,    const float* __restrict__ W,
    const float* __restrict__ bias, int K, int Ntot,
    const int*  __restrict__ list, const int* __restrict__ cnt, int maxRows,
    const float* __restrict__ cw,  const int* __restrict__ offs,
    u16* __restrict__ outBf, float* __restrict__ outF)
{
    constexpr bool EXPERT = (MODE == 0 || MODE == 2);
    __shared__ __align__(16) u16 lA[2][128][40];  // +8 pad: conflict-free frags
    __shared__ __align__(16) u16 lB[2][128][40];  // W tile transposed [n][k]

    const int e      = blockIdx.z;
    const int count  = EXPERT ? cnt[e] : maxRows;
    const int m_base = blockIdx.x * 128;
    if (m_base >= count) return;
    const int n_base = blockIdx.y * 128;
    const int hoff   = EXPERT ? offs[e] : 0;

    const float* We = W    + (size_t)e * K * Ntot;
    const float* be = bias + (size_t)e * Ntot;
    const int*   le = list + (size_t)e * maxRows;
    const float* ce = cw   + (size_t)e * maxRows;

    const int tid = threadIdx.x;
    // ---- A staging: 2 threads/row, 32 bytes each, pure async bf16 copy ----
    const int arow = tid >> 1, acol = (tid & 1) * 16;
    int r  = m_base + arow;
    int rc = (r < count) ? r : (count - 1);
    int gi;
    if      (MODE == 0) gi = le[rc];
    else if (MODE == 2) gi = hoff + rc;
    else                gi = r;
    const u16* Ag = A + (size_t)gi * K;

    // ---- B staging: thread owns 2 K-rows x 8 cols; pk-cvt -> ds_store_b32 ----
    const int bk2 = (tid >> 4) * 2;        // K row pair: 0,2,...,30
    const int bn  = (tid & 15) * 8;        // col base: 0..120

    // ---- wave fragment mapping ----
    const int lane = tid & 31, wv = tid >> 5;
    const int wm   = (wv & 3) * 32;        // wave M offset
    const int wn   = (wv >> 2) * 64;       // wave N offset
    const int frow = lane & 15;
    const int akb  = (lane >> 4) * 8;      // A K-base (ISA 16-bit A layout)
    const int bkb  = (lane >> 4) * 8;      // B K-base (ISA 16-bit B layout)
    const int fcol = lane & 15;

    f32x8 acc[2][4];
#pragma unroll
    for (int mi = 0; mi < 2; mi++)
#pragma unroll
        for (int j = 0; j < 4; j++)
#pragma unroll
            for (int q = 0; q < 8; q++) acc[mi][j][q] = 0.f;

    const int nk = K / 32;

    // ---------------- prologue: tile 0 into buffer 0 ----------------
    async_cp16(Ag + acol,     &lA[0][arow][acol]);
    async_cp16(Ag + acol + 8, &lA[0][arow][acol + 8]);
    {
        const float* Wp = We + (size_t)bk2 * Ntot + n_base + bn;
        float4 r0a = *(const float4*)(Wp);
        float4 r0b = *(const float4*)(Wp + 4);
        float4 r1a = *(const float4*)(Wp + Ntot);
        float4 r1b = *(const float4*)(Wp + Ntot + 4);
        float ra[8] = {r0a.x, r0a.y, r0a.z, r0a.w, r0b.x, r0b.y, r0b.z, r0b.w};
        float rb[8] = {r1a.x, r1a.y, r1a.z, r1a.w, r1b.x, r1b.y, r1b.z, r1b.w};
#pragma unroll
        for (int j = 0; j < 8; j++)
            *(u32*)&lB[0][bn + j][bk2] = pack_bf16(ra[j], rb[j]);
    }
    ASYNC_WAIT0();
    __syncthreads();

    // ---------------- main pipelined loop ----------------
    for (int kt = 0; kt < nk; ++kt) {
        const int cur = kt & 1, nxt = cur ^ 1;
        const bool more = (kt + 1) < nk;
        float ra[8], rb[8];
        if (more) {
            const int k1 = (kt + 1) * 32;
            async_cp16(Ag + k1 + acol,     &lA[nxt][arow][acol]);
            async_cp16(Ag + k1 + acol + 8, &lA[nxt][arow][acol + 8]);
            const float* Wp = We + (size_t)(k1 + bk2) * Ntot + n_base + bn;
            float4 r0a = *(const float4*)(Wp);
            float4 r0b = *(const float4*)(Wp + 4);
            float4 r1a = *(const float4*)(Wp + Ntot);
            float4 r1b = *(const float4*)(Wp + Ntot + 4);
            ra[0]=r0a.x; ra[1]=r0a.y; ra[2]=r0a.z; ra[3]=r0a.w;
            ra[4]=r0b.x; ra[5]=r0b.y; ra[6]=r0b.z; ra[7]=r0b.w;
            rb[0]=r1a.x; rb[1]=r1a.y; rb[2]=r1a.z; rb[3]=r1a.w;
            rb[4]=r1b.x; rb[5]=r1b.y; rb[6]=r1b.z; rb[7]=r1b.w;
            if (kt + 2 < nk)
                __builtin_prefetch((const void*)(Wp + (size_t)32 * Ntot), 0, 1);
        }

        // ---- compute current buffer ----
        FragBF a0, a1;
        a0.q[0] = *(const uint4*)&lA[cur][wm + frow][akb];
        a0.q[1] = *(const uint4*)&lA[cur][wm + frow][akb + 16];
        a1.q[0] = *(const uint4*)&lA[cur][wm + 16 + frow][akb];
        a1.q[1] = *(const uint4*)&lA[cur][wm + 16 + frow][akb + 16];
#pragma unroll
        for (int j = 0; j < 4; j++) {
            FragBF b;
            b.q[0] = *(const uint4*)&lB[cur][wn + j * 16 + fcol][bkb];
            b.q[1] = *(const uint4*)&lB[cur][wn + j * 16 + fcol][bkb + 16];
            acc[0][j] = __builtin_amdgcn_wmma_f32_16x16x32_bf16(
                false, a0.v, false, b.v, (short)0, acc[0][j], false, false);
            acc[1][j] = __builtin_amdgcn_wmma_f32_16x16x32_bf16(
                false, a1.v, false, b.v, (short)0, acc[1][j], false, false);
        }

        if (more) {
#pragma unroll
            for (int j = 0; j < 8; j++)
                *(u32*)&lB[nxt][bn + j][bk2] = pack_bf16(ra[j], rb[j]);
            ASYNC_WAIT0();
        }
        __syncthreads();
    }

    // ---------------- epilogue ----------------
    float bv[4];
#pragma unroll
    for (int j = 0; j < 4; j++) bv[j] = be[n_base + wn + j * 16 + fcol];

#pragma unroll
    for (int mi = 0; mi < 2; mi++)
#pragma unroll
    for (int q = 0; q < 8; q++) {
        int row_t = wm + mi * 16 + (lane >> 4) * 8 + q;
        int rowg  = m_base + row_t;
        bool ok = (rowg < count);
        if (MODE == 0 || MODE == 1) {
            if (ok) {
                size_t ro = (size_t)((MODE == 0) ? (hoff + rowg) : rowg) * Ntot;
#pragma unroll
                for (int j = 0; j < 4; j++) {
                    float v = gelu_exact(acc[mi][j][q] + bv[j]);
                    outBf[ro + n_base + wn + j * 16 + fcol] = f2bf(v);
                }
            }
        } else if (MODE == 2) {
            if (ok) {
                int   tok = le[rowg];
                float s   = ce[rowg];
                size_t ro = (size_t)tok * Ntot;
#pragma unroll
                for (int j = 0; j < 4; j++)
                    atomicAdd(&outF[ro + n_base + wn + j * 16 + fcol],
                              s * (acc[mi][j][q] + bv[j]));
            }
        } else { // MODE 3
            size_t ro = (size_t)rowg * Ntot;
#pragma unroll
            for (int j = 0; j < 4; j++)
                outF[ro + n_base + wn + j * 16 + fcol] = acc[mi][j][q] + bv[j];
        }
    }
}

// ------------------------------ launcher ------------------------------------
extern "C" void kernel_launch(void* const* d_in, const int* in_sizes, int n_in,
                              void* d_out, int out_size, void* d_ws, size_t ws_size,
                              hipStream_t stream) {
    const float* x    = (const float*)d_in[0];
    const float* gw   = (const float*)d_in[1];
    const float* gb   = (const float*)d_in[2];
    const float* w1   = (const float*)d_in[3];
    const float* b1   = (const float*)d_in[4];
    const float* w2   = (const float*)d_in[5];
    const float* b2   = (const float*)d_in[6];
    const float* sw1  = (const float*)d_in[7];
    const float* sb1  = (const float*)d_in[8];
    const float* sw2  = (const float*)d_in[9];
    const float* sb2  = (const float*)d_in[10];
    float* out = (float*)d_out;

    char* ws = (char*)d_ws;
    constexpr size_t OFF_XBF  = 0;                                    // N*D bf16
    constexpr size_t OFF_SH   = OFF_XBF + (size_t)N_TOK * DIM * 2;    // N*H bf16
    constexpr size_t OFF_H    = OFF_SH  + (size_t)N_TOK * HID * 2;    // 2N*H bf16
    constexpr size_t OFF_LIST = OFF_H   + (size_t)2 * N_TOK * HID * 2;
    constexpr size_t OFF_CW   = OFF_LIST + (size_t)NE * N_TOK * 4;
    constexpr size_t OFF_CTR  = OFF_CW   + (size_t)NE * N_TOK * 4;

    u16*   x_bf  = (u16*)(ws + OFF_XBF);
    u16*   sh_bf = (u16*)(ws + OFF_SH);
    u16*   h_bf  = (u16*)(ws + OFF_H);
    int*   list  = (int*)(ws + OFF_LIST);
    float* cw    = (float*)(ws + OFF_CW);
    int*   cnt   = (int*)(ws + OFF_CTR);
    float* imp   = (float*)(cnt + NE);
    float* loadf = imp + NE;
    int*   offs  = (int*)(loadf + NE);

    moe_zero_ctr<<<1, 32, 0, stream>>>(cnt);
    moe_cvt_bf16<<<(N_TOK * DIM) / 1024, 256, 0, stream>>>(x, x_bf);
    moe_gating<<<N_TOK / 8, 256, 0, stream>>>(x, gw, gb, cnt, imp, loadf, list, cw);
    moe_finalize<<<1, 32, 0, stream>>>(cnt, imp, loadf, offs, out);

    // shared expert: FFN1 then FFN2 (FFN2 stores -> initializes d_out)
    moe_gemm<1><<<dim3(N_TOK / 128, HID / 128, 1), 256, 0, stream>>>(
        x_bf, sw1, sb1, DIM, HID, list, cnt, N_TOK, cw, offs, sh_bf, nullptr);
    moe_gemm<3><<<dim3(N_TOK / 128, DIM / 128, 1), 256, 0, stream>>>(
        sh_bf, sw2, sb2, HID, DIM, list, cnt, N_TOK, cw, offs, nullptr, out);

    // routed experts: FFN1 (gathered GELU -> compact h slab), FFN2 (scaled atomics)
    moe_gemm<0><<<dim3(N_TOK / 128, HID / 128, NE), 256, 0, stream>>>(
        x_bf, w1, b1, DIM, HID, list, cnt, N_TOK, cw, offs, h_bf, nullptr);
    moe_gemm<2><<<dim3(N_TOK / 128, DIM / 128, NE), 256, 0, stream>>>(
        h_bf, w2, b2, HID, DIM, list, cnt, N_TOK, cw, offs, nullptr, out);
}
